// GraphAttentionalLayer_43233140801869
// MI455X (gfx1250) — compile-verified, matrix operand order checked by reading, not run on hardware
//
#include <hip/hip_runtime.h>

// ---------------------------------------------------------------------------
// GAT layer, fused flash-style for MI455X (gfx1250, wave32, WMMA + TDM).
//   B=8, N=2048, D=128  (hardcoded per reference)
//   hp projection  : v_wmma_f32_16x16x32_f16 (f16 in, f32 accum)
//   alpha @ hp     : v_wmma_f32_16x16x32_f16 fused with softmax; alpha never
//                    hits memory (saves 134 MB of HBM traffic).
//   hp chunk stage : tensor_load_to_lds (TDM) double-buffered, with HW pad
//                    giving an 80B LDS row pitch (conflict-free ds_load_b128).
//   lrelu monotone => exact row max = lrelu(el_i + max_j er_j): one-pass
//   softmax, no accumulator rescale.
// ---------------------------------------------------------------------------

typedef __attribute__((ext_vector_type(16))) _Float16     v16h;
typedef __attribute__((ext_vector_type(8)))  float        v8f;
typedef __attribute__((ext_vector_type(4)))  unsigned int v4u;
typedef __attribute__((ext_vector_type(8)))  int          v8i;
typedef __attribute__((ext_vector_type(4)))  int          v4i;

#define BATCH 8
#define NSEQ  2048
#define DDIM  128
#define NEG_SLOPE 0.2f
#define LDS_PITCH 40   // halves per row: 64B payload + 16B pad = 80B

union V16HU { v16h v; float4 f4[2]; };

__device__ __forceinline__ void pack8(v16h& v, int base, float4 x, float4 y) {
    v[base + 0] = (_Float16)x.x; v[base + 1] = (_Float16)x.y;
    v[base + 2] = (_Float16)x.z; v[base + 3] = (_Float16)x.w;
    v[base + 4] = (_Float16)y.x; v[base + 5] = (_Float16)y.y;
    v[base + 6] = (_Float16)y.z; v[base + 7] = (_Float16)y.w;
}

// ---------------------------------------------------------------------------
// TDM: DMA a 32(j) x 128(d) f16 tile of hpT[b] (row-major [d][j], row stride
// 2048 elems) into LDS with 16B pad every 64B -> 80B pitch.  D# built per
// CDNA5 ISA; all fields wave-uniform -> SGPRs.
// amdgpu-toolchain (clang-23) 6-arg builtin form.
// ---------------------------------------------------------------------------
__device__ __forceinline__ void tdm_load_tile(unsigned lds_addr, const _Float16* gaddr)
{
    unsigned long long ga = (unsigned long long)gaddr;
    v4u g0;
    g0.x = 1u;                                   // count=1, user descriptor
    g0.y = lds_addr;                             // bits 63:32  lds_addr
    g0.z = (unsigned)(ga & 0xFFFFFFFFu);         // bits 95:64  global_addr lo
    g0.w = (unsigned)((ga >> 32) & 0x01FFFFFFu)  // bits 120:96 global_addr hi
         | (2u << 30);                           // bits 127:126 type=2 (image)

    v8i g1;
    g1[0] = (1 << 16)          // data_size = 1 (2 bytes)
          | (1 << 20)          // pad_enable
          | (3 << 22)          // pad_interval: code 3 -> 16 DWORDs (64B)
          | (3 << 25);         // pad_amount:   code 3 -> 4 DWORDs (16B)
    g1[1] = (int)((NSEQ & 0xFFFF) << 16);        // tensor_dim0 lo -> bits 63:48
    g1[2] = (int)(((NSEQ >> 16) & 0xFFFF)        // tensor_dim0 hi
          | ((DDIM & 0xFFFF) << 16));            // tensor_dim1 lo
    g1[3] = (int)(((DDIM >> 16) & 0xFFFF)        // tensor_dim1 hi
          | (32 << 16));                         // tile_dim0 = 32 (j)
    g1[4] = (int)(DDIM);                         // tile_dim1 = 128 (d), tile_dim2=0
    g1[5] = NSEQ;                                // tensor_dim0_stride lo (elems)
    g1[6] = 0;                                   // stride hi, dim1_stride lo
    g1[7] = 0;

    v4i z4 = {0, 0, 0, 0};                       // 2D tensor: groups 2/3 unused
    v8i z8 = {0, 0, 0, 0, 0, 0, 0, 0};
    __builtin_amdgcn_tensor_load_to_lds(g0, g1, z4, z4, z8, 0);
}

// ---------------------------------------------------------------------------
// Kernel 1: hpT[b][e][n] = sum_d W[e][d] * h[b][n][d]   (f16 output, transposed)
// ---------------------------------------------------------------------------
__global__ __launch_bounds__(256)
void gat_project_kernel(const float* __restrict__ h, const float* __restrict__ W,
                        _Float16* __restrict__ hpT)
{
    const int b    = blockIdx.y;
    const int n0   = blockIdx.x * 128;
    const int wave = threadIdx.x >> 5;
    const int lane = threadIdx.x & 31;
    const int l16  = lane & 15;
    const int hi   = lane >> 4;
    const int e0   = wave * 16;

    const float* __restrict__ hb = h + (size_t)b * NSEQ * DDIM;

    v8f acc[8];
#pragma unroll
    for (int t = 0; t < 8; ++t)
#pragma unroll
        for (int r = 0; r < 8; ++r) acc[t][r] = 0.0f;

#pragma unroll
    for (int kc = 0; kc < DDIM; kc += 32) {
        const int koff = kc + hi * 8;
        const float* wrow = W + (size_t)(e0 + l16) * DDIM + koff;
        float4 a0 = *(const float4*)(wrow);
        float4 a1 = *(const float4*)(wrow + 4);
        float4 a2 = *(const float4*)(wrow + 16);
        float4 a3 = *(const float4*)(wrow + 20);
        v16h A;
        pack8(A, 0, a0, a1);
        pack8(A, 8, a2, a3);

#pragma unroll
        for (int t = 0; t < 8; ++t) {
            const float* hrow = hb + (size_t)(n0 + t * 16 + l16) * DDIM + koff;
            float4 b0 = *(const float4*)(hrow);
            float4 b1 = *(const float4*)(hrow + 4);
            float4 b2 = *(const float4*)(hrow + 16);
            float4 b3 = *(const float4*)(hrow + 20);
            v16h Bf;
            pack8(Bf, 0, b0, b1);
            pack8(Bf, 8, b2, b3);
            acc[t] = __builtin_amdgcn_wmma_f32_16x16x32_f16(
                false, A, false, Bf, (short)0, acc[t], false, false);
        }
    }

#pragma unroll
    for (int t = 0; t < 8; ++t)
#pragma unroll
        for (int r = 0; r < 8; ++r)
            hpT[((size_t)b * DDIM + e0 + r + hi * 8) * NSEQ + n0 + t * 16 + l16] =
                (_Float16)acc[t][r];
}

// ---------------------------------------------------------------------------
// Kernel 2: el[b,n] = sum_e hpT[b][e][n]*a[e]; er likewise with a[128+e].
// ---------------------------------------------------------------------------
__global__ __launch_bounds__(256)
void gat_scores_kernel(const _Float16* __restrict__ hpT, const float* __restrict__ a,
                       float* __restrict__ el, float* __restrict__ er)
{
    const int gid = blockIdx.x * 256 + threadIdx.x;
    const int b   = gid >> 11;
    const int n   = gid & (NSEQ - 1);
    const _Float16* __restrict__ base = hpT + (size_t)b * DDIM * NSEQ + n;
    float sl = 0.0f, sr = 0.0f;
#pragma unroll 8
    for (int e = 0; e < DDIM; ++e) {
        float v = (float)base[(size_t)e * NSEQ];
        sl += v * a[e];
        sr += v * a[DDIM + e];
    }
    el[gid] = sl;
    er[gid] = sr;
}

// ---------------------------------------------------------------------------
// Kernel 3: fused softmax(lrelu(el_i + er_j)) @ hp with TDM double-buffering.
// Block: 256 thr = 8 waves; 128 i-rows x full 128 d per block.
// ---------------------------------------------------------------------------
__global__ __launch_bounds__(256)
void gat_attn_kernel(const _Float16* __restrict__ hpT, const float* __restrict__ el,
                     const float* __restrict__ er, float* __restrict__ out)
{
    __shared__ float    er_s[NSEQ];                  // 8 KB
    __shared__ _Float16 hp_s[2][128 * LDS_PITCH];    // 2 x 10 KB, 80B pitch
    __shared__ float    l_s[128];
    __shared__ float    mx_s[8];

    const int b    = blockIdx.y;
    const int i0   = blockIdx.x * 128;
    const int tid  = threadIdx.x;
    const int wave = tid >> 5;
    const int lane = tid & 31;
    const int l16  = lane & 15;
    const int hi   = lane >> 4;

    // --- stage er[b] into LDS, compute batch max(er) ---
    const float* __restrict__ erb = er + (size_t)b * NSEQ;
    float m = -1e30f;
    for (int j = tid; j < NSEQ; j += 256) {
        float v = erb[j];
        er_s[j] = v;
        m = fmaxf(m, v);
    }
#pragma unroll
    for (int off = 16; off > 0; off >>= 1) m = fmaxf(m, __shfl_xor(m, off, 32));
    if (lane == 0) mx_s[wave] = m;

    const _Float16* __restrict__ hpb = hpT + (size_t)b * DDIM * NSEQ;
    const unsigned lds0 = (unsigned)(unsigned long long)&hp_s[0][0];
    const unsigned lds1 = (unsigned)(unsigned long long)&hp_s[1][0];

    // prologue: kick DMA of first chunk while we finish the max reduction
    if (wave == 0) tdm_load_tile(lds0, hpb);

    __syncthreads();
    float ermax = mx_s[0];
#pragma unroll
    for (int w = 1; w < 8; ++w) ermax = fmaxf(ermax, mx_s[w]);

    const float ell = el[(size_t)b * NSEQ + i0 + wave * 16 + l16];
    float mrow = ell + ermax;
    mrow = fmaxf(mrow, NEG_SLOPE * mrow);

    v8f acc[8];
#pragma unroll
    for (int t = 0; t < 8; ++t)
#pragma unroll
        for (int r = 0; r < 8; ++r) acc[t][r] = 0.0f;
    float lsum = 0.0f;

    int cur = 0;
    for (int jb = 0; jb < NSEQ; jb += 32) {
        if (wave == 0) __builtin_amdgcn_s_wait_tensorcnt(0);  // chunk `cur` landed
        __syncthreads();                                      // visible to all waves
        if (wave == 0 && jb + 32 < NSEQ)                      // prefetch next chunk
            tdm_load_tile(cur ? lds0 : lds1, hpb + jb + 32);

        const _Float16* __restrict__ hbuf = hp_s[cur];

        // --- A fragment: p = exp(lrelu(ell + er_j) - mrow) ---
        const int koff = hi * 8;
        float4 u0 = *(const float4*)(er_s + jb + koff);
        float4 u1 = *(const float4*)(er_s + jb + koff + 4);
        float4 u2 = *(const float4*)(er_s + jb + koff + 16);
        float4 u3 = *(const float4*)(er_s + jb + koff + 20);
        v16h A;
        {
            float q[16] = {u0.x, u0.y, u0.z, u0.w, u1.x, u1.y, u1.z, u1.w,
                           u2.x, u2.y, u2.z, u2.w, u3.x, u3.y, u3.z, u3.w};
#pragma unroll
            for (int k = 0; k < 16; ++k) {
                float s_ = ell + q[k];
                s_ = fmaxf(s_, NEG_SLOPE * s_);
                float p_ = __expf(s_ - mrow);
                lsum += p_;
                A[k] = (_Float16)p_;
            }
        }

        // --- 8 d-tiles: conflict-free b128 B-fragment loads from LDS ---
#pragma unroll
        for (int t = 0; t < 8; ++t) {
            const _Float16* row = hbuf + (t * 16 + l16) * LDS_PITCH + koff;
            V16HU Bu;
            Bu.f4[0] = *(const float4*)(row);
            Bu.f4[1] = *(const float4*)(row + 16);
            acc[t] = __builtin_amdgcn_wmma_f32_16x16x32_f16(
                false, A, false, Bu.v, (short)0, acc[t], false, false);
        }
        cur ^= 1;
    }

    // --- row sums: lane & lane^16 cover complementary j sets of same row ---
    float rowsum = lsum + __shfl_xor(lsum, 16, 32);
    if (hi == 0) l_s[wave * 16 + l16] = rowsum;
    __syncthreads();

    float inv8[8];
#pragma unroll
    for (int r = 0; r < 8; ++r) inv8[r] = 1.0f / l_s[wave * 16 + r + hi * 8];

    float* __restrict__ ob = out + ((size_t)b * NSEQ + i0 + wave * 16) * DDIM;
#pragma unroll
    for (int t = 0; t < 8; ++t)
#pragma unroll
        for (int r = 0; r < 8; ++r)
            ob[(size_t)(r + hi * 8) * DDIM + t * 16 + l16] = acc[t][r] * inv8[r];
}

// ---------------------------------------------------------------------------
extern "C" void kernel_launch(void* const* d_in, const int* in_sizes, int n_in,
                              void* d_out, int out_size, void* d_ws, size_t ws_size,
                              hipStream_t stream)
{
    const float* h = (const float*)d_in[0];   // (8, 2048, 128) f32
    const float* W = (const float*)d_in[1];   // (128, 128)     f32
    const float* a = (const float*)d_in[2];   // (256,)         f32
    float* out = (float*)d_out;               // (8, 2048, 128) f32

    char* ws = (char*)d_ws;
    _Float16* hpT = (_Float16*)ws;                                  // 4 MB
    float*    el  = (float*)(ws + (size_t)BATCH * DDIM * NSEQ * 2); // 64 KB
    float*    er  = el + (size_t)BATCH * NSEQ;                      // 64 KB

    gat_project_kernel<<<dim3(NSEQ / 128, BATCH), 256, 0, stream>>>(h, W, hpT);
    gat_scores_kernel<<<dim3((BATCH * NSEQ) / 256), 256, 0, stream>>>(hpT, a, el, er);
    gat_attn_kernel<<<dim3(NSEQ / 128, BATCH), 256, 0, stream>>>(hpT, el, er, out);
}
